// RNN_40699110096897
// MI455X (gfx1250) — compile-verified, matrix operand order checked by reading, not run on hardware
//
#include <hip/hip_runtime.h>

typedef __attribute__((ext_vector_type(16))) _Float16 v16h;
typedef __attribute__((ext_vector_type(8)))  _Float16 v8h;
typedef __attribute__((ext_vector_type(8)))  float    v8f;

#define T_STEPS 4096
#define BATCH   64
#define DDIM    128
#define LDIM    128

#define CAT16(lo, hi) __builtin_shufflevector((lo), (hi), 0,1,2,3,4,5,6,7,8,9,10,11,12,13,14,15)

__device__ __forceinline__ float fast_tanh(float v) {
#if __has_builtin(__builtin_amdgcn_tanhf)
  return __builtin_amdgcn_tanhf(v);   // hardware V_TANH_F32 transcendental
#else
  return tanhf(v);
#endif
}

// ---------------------------------------------------------------------------
// Phase 1: xp[t*B + b, :] = x[t,b,:] @ Wx + bx        (bandwidth bound)
// 262144 x 128 x 128 GEMM via v_wmma_f32_16x16x32_f16.
// Wx is pre-swizzled once per block into exact WMMA B-fragment layout in LDS;
// the compiler then promotes all 32 B fragments to (high) VGPRs for the loop.
// ---------------------------------------------------------------------------
__global__ __launch_bounds__(256) void rnn_xproj(const float* __restrict__ x,
                                                 const float* __restrict__ Wx,
                                                 const float* __restrict__ bx,
                                                 float* __restrict__ xp) {
  __shared__ _Float16 bfrag[4 * 8 * 32 * 16];  // 32 KB, WMMA B fragments of Wx
  const int tid = threadIdx.x;

  // Swizzle-fill Wx (f32 -> f16) into B-fragment layout.
  #pragma unroll 4
  for (int i = 0; i < 64; ++i) {
    int f    = i * 256 + tid;              // 16384 halves total
    int e    = f & 15;
    int lane = (f >> 4) & 31;
    int nt   = (f >> 9) & 7;
    int kc   = f >> 12;
    int n = nt * 16 + (lane & 15);
    int k = kc * 32 + (e & 7) + ((e >> 3) << 4) + ((lane >> 4) << 3);
    bfrag[f] = (_Float16)Wx[k * LDIM + n];
  }
  __syncthreads();

  const int wave = tid >> 5;
  const int lane = tid & 31;

  // Per-lane bias for each N tile (C/D layout: N = nt*16 + (lane&15)).
  float biasn[8];
  #pragma unroll
  for (int nt = 0; nt < 8; ++nt) biasn[nt] = bx[nt * 16 + (lane & 15)];

  const int totalStripes = (T_STEPS * BATCH) / 16;  // 16384 row stripes of 16
  for (int s0 = blockIdx.x * 8; s0 < totalStripes; s0 += gridDim.x * 8) {
    const int s  = s0 + wave;
    const int m0 = s * 16;
    const int r  = m0 + (lane & 15);

    v8f c[8];
    #pragma unroll
    for (int nt = 0; nt < 8; ++nt)
      #pragma unroll
      for (int v = 0; v < 8; ++v) c[nt][v] = biasn[nt];  // bias folded into acc

    #pragma unroll
    for (int kc = 0; kc < 4; ++kc) {
      const int kbase = kc * 32 + ((lane >> 4) << 3);
      const float4* px = (const float4*)(x + (size_t)r * DDIM + kbase);
      float4 f0 = px[0], f1 = px[1];   // k+0..7
      float4 f2 = px[4], f3 = px[5];   // k+16..23
      float fl[8], fh[8];
      *(float4*)&fl[0] = f0; *(float4*)&fl[4] = f1;
      *(float4*)&fh[0] = f2; *(float4*)&fh[4] = f3;
      v16h a;
      #pragma unroll
      for (int j = 0; j < 8; ++j) {
        a[j]     = (_Float16)fl[j];
        a[j + 8] = (_Float16)fh[j];
      }
      #pragma unroll
      for (int nt = 0; nt < 8; ++nt) {
        const v8h* pb = (const v8h*)&bfrag[((kc * 8 + nt) * 32 + lane) * 16];
        v16h b = CAT16(pb[0], pb[1]);
        c[nt] = __builtin_amdgcn_wmma_f32_16x16x32_f16(
            false, a, false, b, (short)0, c[nt], false, false);
      }
    }

    // Store f32 xp. C layout: row = m0 + (lane>=16 ? 8:0) + v, col = nt*16+(lane&15)
    const int rowbase = m0 + ((lane >> 4) << 3);
    #pragma unroll
    for (int nt = 0; nt < 8; ++nt) {
      const int n = nt * 16 + (lane & 15);
      #pragma unroll
      for (int v = 0; v < 8; ++v)
        xp[(size_t)(rowbase + v) * LDIM + n] = c[nt][v];
    }
  }
}

// ---------------------------------------------------------------------------
// Phase 2: sequential scan h = tanh(xp_t + h @ Wh + bh)   (latency bound)
// Batch rows are independent -> 4 workgroups, each owns 16 batch rows.
// 8 waves per WG, one 16x16 N-tile each; Wh pre-swizzled into WMMA B frags
// (resident in VGPRs), h resident in LDS (f16, double buffered).
// Critical-path tuning: xp(t+1) software-pipelined into registers, WMMA
// accumulation split into two independent 2-deep chains, hardware tanh.
// ---------------------------------------------------------------------------
__global__ __launch_bounds__(256) void rnn_scan(const float* __restrict__ xp,
                                                const float* __restrict__ h0,
                                                const float* __restrict__ Wh,
                                                const float* __restrict__ bh,
                                                float* __restrict__ out) {
  __shared__ _Float16 wfrag[4 * 8 * 32 * 16];     // 32 KB: Wh B fragments
  __shared__ _Float16 hbuf[2][16][LDIM];          //  8 KB: double-buffered h
  const int tid = threadIdx.x;
  const int mt  = blockIdx.x;                     // batch slice: rows mt*16..+15

  // Swizzle-fill Wh into B-fragment layout (once).
  #pragma unroll 4
  for (int i = 0; i < 64; ++i) {
    int f    = i * 256 + tid;
    int e    = f & 15;
    int lane = (f >> 4) & 31;
    int nt   = (f >> 9) & 7;
    int kc   = f >> 12;
    int n = nt * 16 + (lane & 15);
    int k = kc * 32 + (e & 7) + ((e >> 3) << 4) + ((lane >> 4) << 3);
    wfrag[f] = (_Float16)Wh[k * LDIM + n];
  }
  // Load initial h slice (f32 -> f16).
  #pragma unroll
  for (int i = 0; i < 8; ++i) {
    int f  = i * 256 + tid;                      // 2048 halves
    int lr = f >> 7;
    int n  = f & 127;
    hbuf[0][lr][n] = (_Float16)h0[(mt * 16 + lr) * LDIM + n];
  }
  __syncthreads();

  const int wave   = tid >> 5;
  const int lane   = tid & 31;
  const int nt     = wave;                        // N tile owned by this wave
  const int n      = nt * 16 + (lane & 15);       // C/D column
  const int lrbase = (lane >> 4) << 3;            // 0 or 8 (C/D row group)
  const float biasn = bh[n];

  // Preload B fragments for this wave's N tile (never change): 32 VGPRs.
  v16h bf[4];
  #pragma unroll
  for (int kc = 0; kc < 4; ++kc) {
    const v8h* pb = (const v8h*)&wfrag[((kc * 8 + nt) * 32 + lane) * 16];
    bf[kc] = CAT16(pb[0], pb[1]);
  }

  // Software-pipelined xp tile: preload t=0.
  float xv[8];
  {
    const size_t rowg0 = (size_t)mt * 16 + lrbase;
    #pragma unroll
    for (int v = 0; v < 8; ++v) xv[v] = xp[(rowg0 + v) * LDIM + n];
  }

  for (int t = 0; t < T_STEPS; ++t) {
    const int cur = t & 1, nxt = cur ^ 1;
    const size_t rowg = (size_t)t * BATCH + mt * 16 + lrbase;

    // Two independent accumulation chains halve dependent-WMMA depth.
    v8f acc0, acc1;
    #pragma unroll
    for (int v = 0; v < 8; ++v) { acc0[v] = xv[v] + biasn; acc1[v] = 0.0f; }

    // Kick off next timestep's xp loads immediately (consumed next iter,
    // so their latency overlaps this whole step incl. the barrier).
    if (t + 1 < T_STEPS) {
      #pragma unroll
      for (int v = 0; v < 8; ++v)
        xv[v] = xp[(rowg + BATCH + v) * LDIM + n];
    }

    // h @ Wh : A fragments straight from the resident h buffer.
    #pragma unroll
    for (int kc = 0; kc < 4; ++kc) {
      const int kbase = kc * 32 + lrbase;         // (lane>=16 ? +8 : 0)
      const v8h alo = *(const v8h*)&hbuf[cur][lane & 15][kbase];
      const v8h ahi = *(const v8h*)&hbuf[cur][lane & 15][kbase + 16];
      v16h a = CAT16(alo, ahi);
      if (kc & 1)
        acc1 = __builtin_amdgcn_wmma_f32_16x16x32_f16(
            false, a, false, bf[kc], (short)0, acc1, false, false);
      else
        acc0 = __builtin_amdgcn_wmma_f32_16x16x32_f16(
            false, a, false, bf[kc], (short)0, acc0, false, false);
    }

    // Merge chains, tanh, emit f32 output, stage f16 h for next step.
    #pragma unroll
    for (int v = 0; v < 8; ++v) {
      float hn = fast_tanh(acc0[v] + acc1[v]);
      out[(rowg + v) * LDIM + n] = hn;
      hbuf[nxt][lrbase + v][n] = (_Float16)hn;
    }
    __syncthreads();  // h buffer swap barrier
  }
}

// ---------------------------------------------------------------------------
extern "C" void kernel_launch(void* const* d_in, const int* in_sizes, int n_in,
                              void* d_out, int out_size, void* d_ws, size_t ws_size,
                              hipStream_t stream) {
  const float* x  = (const float*)d_in[0];   // [T,B,D]
  const float* h  = (const float*)d_in[1];   // [B,L]
  const float* Wx = (const float*)d_in[2];   // [D,L]
  const float* bx = (const float*)d_in[3];   // [L]
  const float* Wh = (const float*)d_in[4];   // [L,L]
  const float* bh = (const float*)d_in[5];   // [L]
  float* out = (float*)d_out;                // [T,B,L]
  float* xp  = (float*)d_ws;                 // [T*B,L] f32 = 128 MiB scratch

  rnn_xproj<<<512, 256, 0, stream>>>(x, Wx, bx, xp);
  rnn_scan<<<4, 256, 0, stream>>>(xp, h, Wh, bh, out);
}
